// MultiheadAttention_84026740179276
// MI455X (gfx1250) — matrix operand order repaired and checked by reading, not compiled, not run on hardware
//
#include <hip/hip_runtime.h>
#include <hip/hip_bf16.h>

typedef __attribute__((ext_vector_type(16))) __bf16 v16bf;
typedef __attribute__((ext_vector_type(8)))  float  v8f;
typedef __attribute__((ext_vector_type(2)))  __bf16 v2bf;

union BF16Frag { v16bf v; unsigned int u[8]; };

// ---- feature probes -------------------------------------------------------
#if defined(__has_builtin)
#  if __has_builtin(__builtin_amdgcn_cvt_pk_bf16_f32)
#    define USE_PK_CVT 1
#  endif
#  if __has_builtin(__builtin_amdgcn_s_wait_asynccnt)
#    define HAVE_WAIT_ASYNC_BUILTIN 1
#  endif
#  if __has_builtin(__builtin_amdgcn_exp2f)
#    define HAVE_RAW_EXP2 1
#  endif
#  if __has_builtin(__builtin_amdgcn_update_dpp)
#    define HAVE_DPP 1
#  endif
#  if __has_builtin(__builtin_amdgcn_rcpf)
#    define HAVE_RCP 1
#  endif
#endif
#ifndef USE_PK_CVT
#  define USE_PK_CVT 0
#endif
#ifndef HAVE_WAIT_ASYNC_BUILTIN
#  define HAVE_WAIT_ASYNC_BUILTIN 0
#endif
#ifndef HAVE_RAW_EXP2
#  define HAVE_RAW_EXP2 0
#endif
#ifndef HAVE_DPP
#  define HAVE_DPP 0
#endif
#ifndef HAVE_RCP
#  define HAVE_RCP 0
#endif

__device__ inline unsigned short f2bf(float f) {
    unsigned int u = __builtin_bit_cast(unsigned int, f);
    u += 0x7fffu + ((u >> 16) & 1u);           // round-to-nearest-even
    return (unsigned short)(u >> 16);
}
__device__ inline unsigned int pack2(float a, float b) {
#if USE_PK_CVT
    v2bf p = __builtin_amdgcn_cvt_pk_bf16_f32(a, b);
    return __builtin_bit_cast(unsigned int, p);
#else
    return (unsigned int)f2bf(a) | ((unsigned int)f2bf(b) << 16);
#endif
}
__device__ inline float fast_exp2(float x) {   // args are <= 0; flush-to-zero ok
#if HAVE_RAW_EXP2
    return __builtin_amdgcn_exp2f(x);
#else
    return exp2f(x);
#endif
}
__device__ inline float fast_rcp(float x) {
#if HAVE_RCP
    return __builtin_amdgcn_rcpf(x);
#else
    return 1.f / x;
#endif
}

// 16-lane row reduction via DPP row_ror (no LDS round trips)
#if HAVE_DPP
template<int CTRL>
__device__ inline float dpp_max_step(float x) {
    int xi = __builtin_bit_cast(int, x);
    int yi = __builtin_amdgcn_update_dpp(xi, xi, CTRL, 0xF, 0xF, true);
    return fmaxf(x, __builtin_bit_cast(float, yi));
}
__device__ inline float row16_max(float x) {
    x = dpp_max_step<0x121>(x);    // row_ror:1
    x = dpp_max_step<0x122>(x);    // row_ror:2
    x = dpp_max_step<0x124>(x);    // row_ror:4
    x = dpp_max_step<0x128>(x);    // row_ror:8
    return x;
}
#else
__device__ inline float row16_max(float x) {
    for (int off = 1; off < 16; off <<= 1) x = fmaxf(x, __shfl_xor(x, off, 32));
    return x;
}
#endif

// Async 16B global->LDS copy via inline CDNA5 asm (VGLOBAL async encoding,
// GV mode: 64-bit VGPR address, saddr=off). VDST carries the LDS byte
// address; generic shared pointers truncate to the LDS offset (ISA 10.2).
__device__ inline void async_cp16(const void* gptr, void* lptr) {
    unsigned lds_off = (unsigned)(size_t)lptr;
    asm volatile("global_load_async_to_lds_b128 %0, %1, off"
                 :: "v"(lds_off), "v"(gptr) : "memory");
}
__device__ inline void wait_async0() {
#if HAVE_WAIT_ASYNC_BUILTIN
    __builtin_amdgcn_s_wait_asynccnt(0);
#else
    asm volatile("s_wait_asynccnt 0x0" ::: "memory");
#endif
}

// A-fragment dword pattern for a 32-K bf16 row (16 dwords/row):
// lanes 0-15: dwords {0..7}; lanes 16-31: dwords {8..15}
__device__ inline int a_dw(int j, int half) {
    return half * 8 + j;
}

// ---------------------------------------------------------------------------
// Tiled GEMM: C[M,N] = (A[M,K] @ W[N,K]^T + bias[N]) * oscale
// Block = 256 threads (8 waves), tile 128(M) x 128(N), K-step 32.
// Software-pipelined: ping-pong LDS buffers, next tile's global loads are in
// flight while the current tile's 8 WMMAs execute.
// ---------------------------------------------------------------------------
template<bool A_BF16, bool OUT_QKV>
__global__ __launch_bounds__(256) void gemm_wmma(
    const void* __restrict__ Aptr, const float* __restrict__ W,
    const float* __restrict__ bias, void* __restrict__ Out,
    int M, int N, int K, float oscale)
{
    __shared__ unsigned int lA[2][128 * 16];   // 2 x (128 rows x 32 bf16)
    __shared__ unsigned int lB[2][128 * 16];

    const int tid  = threadIdx.x;
    const int wid  = tid >> 5, lane = tid & 31;
    const int half = lane >> 4, lm = lane & 15;
    const int waveM = wid >> 1, waveN = wid & 1;
    const int blockN = blockIdx.x * 128, blockM = blockIdx.y * 128;

    v8f acc[2][4];
#pragma unroll
    for (int i = 0; i < 2; i++)
#pragma unroll
        for (int j = 0; j < 4; j++)
#pragma unroll
            for (int r = 0; r < 8; r++) acc[i][j][r] = 0.f;

    float4 ra[4]; uint4 rh[2]; float4 rw[4];   // register staging

    const int nK = K >> 5;
    // -- prologue: global loads for tile 0 --
    {
        if (A_BF16) {
            const uint4* Ag = (const uint4*)Aptr;
#pragma unroll
            for (int i = 0; i < 2; i++) {
                int idx = tid * 2 + i;
                int row = idx >> 2, q4 = idx & 3;
                rh[i] = Ag[((size_t)(blockM + row) * K + q4 * 8) >> 3];
            }
        } else {
            const float4* Ag = (const float4*)Aptr;
#pragma unroll
            for (int i = 0; i < 4; i++) {
                int idx = tid * 4 + i;
                int row = idx >> 3, cv = idx & 7;
                ra[i] = Ag[((size_t)(blockM + row) * K + cv * 4) >> 2];
            }
        }
        const float4* Wg = (const float4*)W;
#pragma unroll
        for (int i = 0; i < 4; i++) {
            int idx = tid * 4 + i;
            int row = idx >> 3, cv = idx & 7;
            rw[i] = Wg[((size_t)(blockN + row) * K + cv * 4) >> 2];
        }
    }

    for (int kk = 0; kk < nK; ++kk) {
        const int cur = kk & 1;
        // ---- store staged registers into LDS buffer `cur` ----
        if (A_BF16) {
#pragma unroll
            for (int i = 0; i < 2; i++) {
                int idx = tid * 2 + i;
                int row = idx >> 2, q4 = idx & 3;
                lA[cur][row * 16 + q4 * 4 + 0] = rh[i].x;
                lA[cur][row * 16 + q4 * 4 + 1] = rh[i].y;
                lA[cur][row * 16 + q4 * 4 + 2] = rh[i].z;
                lA[cur][row * 16 + q4 * 4 + 3] = rh[i].w;
            }
        } else {
#pragma unroll
            for (int i = 0; i < 4; i++) {
                int idx = tid * 4 + i;
                int row = idx >> 3, cv = idx & 7;
                lA[cur][row * 16 + cv * 2 + 0] = pack2(ra[i].x, ra[i].y);
                lA[cur][row * 16 + cv * 2 + 1] = pack2(ra[i].z, ra[i].w);
            }
        }
#pragma unroll
        for (int i = 0; i < 4; i++) {
            int idx = tid * 4 + i;
            int row = idx >> 3, cv = idx & 7;
            lB[cur][row * 16 + cv * 2 + 0] = pack2(rw[i].x, rw[i].y);
            lB[cur][row * 16 + cv * 2 + 1] = pack2(rw[i].z, rw[i].w);
        }
        __syncthreads();

        // ---- issue next tile's global loads (latency hides under WMMAs) ---
        if (kk + 1 < nK) {
            if (A_BF16) {
                const uint4* Ag = (const uint4*)Aptr;
#pragma unroll
                for (int i = 0; i < 2; i++) {
                    int idx = tid * 2 + i;
                    int row = idx >> 2, q4 = idx & 3;
                    rh[i] = Ag[((size_t)(blockM + row) * K + (kk + 1) * 32 + q4 * 8) >> 3];
                }
            } else {
                const float4* Ag = (const float4*)Aptr;
#pragma unroll
                for (int i = 0; i < 4; i++) {
                    int idx = tid * 4 + i;
                    int row = idx >> 3, cv = idx & 7;
                    ra[i] = Ag[((size_t)(blockM + row) * K + (kk + 1) * 32 + cv * 4) >> 2];
                }
            }
            const float4* Wg = (const float4*)W;
#pragma unroll
            for (int i = 0; i < 4; i++) {
                int idx = tid * 4 + i;
                int row = idx >> 3, cv = idx & 7;
                rw[i] = Wg[((size_t)(blockN + row) * K + (kk + 1) * 32 + cv * 4) >> 2];
            }
            if (kk + 2 < nK)   // prefetch tile kk+2 (global_prefetch_b8)
                __builtin_prefetch((const char*)W +
                    ((size_t)(blockN + (tid >> 1)) * K + (kk + 2) * 32) * 4, 0, 1);
        }

        // ---- compute tile `cur` ----
        BF16Frag a[2], b[4];
#pragma unroll
        for (int mt = 0; mt < 2; mt++) {
            int rowA = waveM * 32 + mt * 16 + lm;
#pragma unroll
            for (int j = 0; j < 8; j++) a[mt].u[j] = lA[cur][rowA * 16 + a_dw(j, half)];
        }
#pragma unroll
        for (int nt = 0; nt < 4; nt++) {
            int n = waveN * 64 + nt * 16 + lm;
#pragma unroll
            for (int j = 0; j < 8; j++) b[nt].u[j] = lB[cur][n * 16 + half * 8 + j];
        }
#pragma unroll
        for (int mt = 0; mt < 2; mt++)
#pragma unroll
            for (int nt = 0; nt < 4; nt++)
                acc[mt][nt] = __builtin_amdgcn_wmma_f32_16x16x32_bf16(
                    false, a[mt].v, false, b[nt].v, (short)0, acc[mt][nt], false, false);
    }

    // ---- epilogue: bias + scale + store ----
#pragma unroll
    for (int mt = 0; mt < 2; mt++) {
#pragma unroll
        for (int nt = 0; nt < 4; nt++) {
            int col = blockN + waveN * 64 + nt * 16 + lm;
            float bv = bias[col];
#pragma unroll
            for (int r = 0; r < 8; r++) {
                int row = blockM + waveM * 32 + mt * 16 + r + 8 * half;
                float v = (acc[mt][nt][r] + bv) * oscale;
                if (OUT_QKV) {
                    int b_ = row >> 11, s = row & 2047;   // S = 2048
                    int h  = col >> 6,  d = col & 63;     // hd = 64
                    ((unsigned short*)Out)[(((size_t)(b_ * 16 + h) * 2048) + s) * 64 + d] = f2bf(v);
                } else {
                    ((float*)Out)[(size_t)row * N + col] = v;
                }
            }
        }
    }
}

// ---------------------------------------------------------------------------
// Flash attention over bf16 Q/K/V in [B,H,S,hd] (S=2048, hd=64).
// Q pre-scaled by (1/sqrt(hd))*log2(e) -> scores live in the exp2 domain.
// Block = 128 threads (4 waves); each block: 64 query rows of one (b,h).
// Double-buffered K (async-to-LDS) and V (register-staged transpose); the
// next key block's loads are in flight during the current block's WMMAs.
// ---------------------------------------------------------------------------
__global__ __launch_bounds__(128) void flash_attn(
    const unsigned short* __restrict__ Q, const unsigned short* __restrict__ Kk,
    const unsigned short* __restrict__ V, unsigned short* __restrict__ Outp)
{
    __shared__ unsigned int sQ[64 * 32];       // 64 x 64 bf16
    __shared__ unsigned int sK[2][32 * 32];    // 2 x (32 x 64 bf16)
    __shared__ unsigned int sVt[2][64 * 16];   // 2 x transposed 64(d) x 32(k)
    __shared__ unsigned int sP[4 * 16 * 16];   // per-wave 16 x 32 bf16 P tiles

    const int tid = threadIdx.x, wid = tid >> 5, lane = tid & 31;
    const int half = lane >> 4, lm = lane & 15;
    const int bh = blockIdx.y;                 // b*16 + h
    const int qbase = blockIdx.x * 64;
    const size_t base = (size_t)bh * 2048 * 64;

    // ---- load Q tile (64x64 bf16 = 512 x 16B) via async copies ----
    {
        const unsigned short* Qg = Q + base + (size_t)qbase * 64;
#pragma unroll
        for (int i = 0; i < 4; i++) {
            int idx = tid * 4 + i;
            async_cp16(Qg + idx * 8, (char*)sQ + idx * 16);
        }
        wait_async0();
    }
    __syncthreads();

    BF16Frag aQ[2];
    {
        int rowA = wid * 16 + lm;
#pragma unroll
        for (int c = 0; c < 2; c++)
#pragma unroll
            for (int j = 0; j < 8; j++)
                aQ[c].u[j] = sQ[rowA * 32 + c * 16 + a_dw(j, half)];
    }

    BF16Frag ones;                             // B = all 1.0bf16 (32x16)
#pragma unroll
    for (int j = 0; j < 8; j++) ones.u[j] = 0x3F803F80u;

    v8f accO[4], lacc;
#pragma unroll
    for (int dt = 0; dt < 4; dt++)
#pragma unroll
        for (int r = 0; r < 8; r++) accO[dt][r] = 0.f;
#pragma unroll
    for (int r = 0; r < 8; r++) lacc[r] = 0.f;
    float mrun = -1e30f;                       // running tile max (log2 domain)

    unsigned int vr[8];                        // register-staged V block

    // ---- prologue: K block 0 async, V block 0 into registers ----
#pragma unroll
    for (int i = 0; i < 2; i++) {
        int idx = tid * 2 + i;
        async_cp16(Kk + base + idx * 8, (char*)sK[0] + idx * 16);
    }
    {
        const unsigned int* Vg = (const unsigned int*)(V + base);
#pragma unroll
        for (int i = 0; i < 8; i++) vr[i] = Vg[tid * 8 + i];
    }

    for (int kb = 0; kb < 2048; kb += 32) {
        const int cur = (kb >> 5) & 1;
        // ---- transpose-store staged V regs into sVt[cur] ----
        {
            unsigned short* vt = (unsigned short*)sVt[cur];
#pragma unroll
            for (int i = 0; i < 8; i++) {
                int idx = tid * 8 + i;
                int k = idx >> 5, dp = idx & 31;
                vt[(2 * dp + 0) * 32 + k] = (unsigned short)(vr[i] & 0xffffu);
                vt[(2 * dp + 1) * 32 + k] = (unsigned short)(vr[i] >> 16);
            }
        }
        wait_async0();                         // K[cur] async copies done
        __syncthreads();

        // ---- issue next block's K async + V register loads ----
        if (kb + 32 < 2048) {
            const unsigned short* Kg = Kk + base + (size_t)(kb + 32) * 64;
#pragma unroll
            for (int i = 0; i < 2; i++) {
                int idx = tid * 2 + i;
                async_cp16(Kg + idx * 8, (char*)sK[cur ^ 1] + idx * 16);
            }
            const unsigned int* Vg = (const unsigned int*)(V + base + (size_t)(kb + 32) * 64);
#pragma unroll
            for (int i = 0; i < 8; i++) vr[i] = Vg[tid * 8 + i];
            if (kb + 64 < 2048) {
                __builtin_prefetch(Kk + base + (size_t)(kb + 64) * 64 + tid * 16, 0, 1);
                __builtin_prefetch(V  + base + (size_t)(kb + 64) * 64 + tid * 16, 0, 1);
            }
        }

        // ---- scores S = Q K^T (16x32 per wave, already log2-scaled) ----
        v8f sc[2];
#pragma unroll
        for (int nt = 0; nt < 2; nt++) {
#pragma unroll
            for (int r = 0; r < 8; r++) sc[nt][r] = 0.f;
#pragma unroll
            for (int c = 0; c < 2; c++) {
                BF16Frag b;
                int n = nt * 16 + lm;
#pragma unroll
                for (int j = 0; j < 8; j++) b.u[j] = sK[cur][n * 32 + c * 16 + half * 8 + j];
                sc[nt] = __builtin_amdgcn_wmma_f32_16x16x32_bf16(
                    false, aQ[c].v, false, b.v, (short)0, sc[nt], false, false);
            }
        }

        // ---- block-wide running max: one DPP row reduction per key block ---
        float tm = -1e30f;
#pragma unroll
        for (int r = 0; r < 8; r++) tm = fmaxf(tm, fmaxf(sc[0][r], sc[1][r]));
        tm = row16_max(tm);
        float mnew  = fmaxf(mrun, tm);
        float alpha = fast_exp2(mrun - mnew);  // v_exp_f32 is native exp2
        mrun = mnew;

        // ---- P = exp2(S - mnew); stash bf16 P in LDS for A-layout reload ---
        unsigned short* pw = (unsigned short*)sP;
#pragma unroll
        for (int r = 0; r < 8; r++) {
            float p0 = fast_exp2(sc[0][r] - mnew);
            float p1 = fast_exp2(sc[1][r] - mnew);
            int rowid = r + 8 * half;
            pw[(wid * 16 + rowid) * 32 +  0 + lm] = f2bf(p0);
            pw[(wid * 16 + rowid) * 32 + 16 + lm] = f2bf(p1);
        }

        BF16Frag aP;
#pragma unroll
        for (int j = 0; j < 8; j++)
            aP.u[j] = sP[(wid * 16 + lm) * 16 + a_dw(j, half)];

        // ---- row sums via WMMA against ones (lands in C layout) ----
        v8f rs;
#pragma unroll
        for (int r = 0; r < 8; r++) rs[r] = 0.f;
        rs = __builtin_amdgcn_wmma_f32_16x16x32_bf16(
            false, aP.v, false, ones.v, (short)0, rs, false, false);
#pragma unroll
        for (int r = 0; r < 8; r++) lacc[r] = lacc[r] * alpha + rs[r];

        // ---- O = O*alpha + P V ----
#pragma unroll
        for (int dt = 0; dt < 4; dt++) {
#pragma unroll
            for (int r = 0; r < 8; r++) accO[dt][r] = accO[dt][r] * alpha;
            BF16Frag b;
            int n = dt * 16 + lm;
#pragma unroll
            for (int j = 0; j < 8; j++) b.u[j] = sVt[cur][n * 16 + half * 8 + j];
            accO[dt] = __builtin_amdgcn_wmma_f32_16x16x32_bf16(
                false, aP.v, false, b.v, (short)0, accO[dt], false, false);
        }
    }

    // ---- epilogue: normalize, store bf16 into concat [B,S,D] layout ----
    const int b_ = bh >> 4, h = bh & 15;
#pragma unroll
    for (int r = 0; r < 8; r++) {
        float rl = fast_rcp(lacc[r]);
        int row = qbase + wid * 16 + r + 8 * half;
#pragma unroll
        for (int dt = 0; dt < 4; dt++) {
            int d = dt * 16 + lm;
            float v = accO[dt][r] * rl;
            Outp[((size_t)(b_ * 2048) + row) * 1024 + h * 64 + d] = f2bf(v);
        }
    }
}

extern "C" void kernel_launch(void* const* d_in, const int* in_sizes, int n_in,
                              void* d_out, int out_size, void* d_ws, size_t ws_size,
                              hipStream_t stream) {
    (void)in_sizes; (void)n_in; (void)out_size; (void)ws_size;
    const float* q_in = (const float*)d_in[0];
    const float* k_in = (const float*)d_in[1];
    const float* v_in = (const float*)d_in[2];
    const float* Wq = (const float*)d_in[3];  const float* bq = (const float*)d_in[4];
    const float* Wk = (const float*)d_in[5];  const float* bk = (const float*)d_in[6];
    const float* Wv = (const float*)d_in[7];  const float* bv = (const float*)d_in[8];
    const float* Wo = (const float*)d_in[9];  const float* bo = (const float*)d_in[10];

    const size_t NEL = (size_t)2 * 2048 * 1024;      // 4,194,304 elements
    unsigned short* qB = (unsigned short*)d_ws;
    unsigned short* kB = qB + NEL;
    unsigned short* vB = kB + NEL;
    unsigned short* aB = vB + NEL;

    const int M = 4096, N = 1024, K = 1024;
    // Q is pre-scaled by (1/sqrt(64)) * log2(e) so attention runs in exp2 domain
    const float qscale = 0.125f * 1.4426950408889634f;
    dim3 gg(N / 128, M / 128);                       // (8, 32)
    gemm_wmma<false, true><<<gg, 256, 0, stream>>>(q_in, Wq, bq, qB, M, N, K, qscale);
    gemm_wmma<false, true><<<gg, 256, 0, stream>>>(k_in, Wk, bk, kB, M, N, K, 1.f);
    gemm_wmma<false, true><<<gg, 256, 0, stream>>>(v_in, Wv, bv, vB, M, N, K, 1.f);

    dim3 ga(2048 / 64, 2 * 16);                      // (32 q-blocks, 32 b*h)
    flash_attn<<<ga, 128, 0, stream>>>(qB, kB, vB, aB);

    gemm_wmma<true, false><<<gg, 256, 0, stream>>>(aB, Wo, bo, d_out, M, N, K, 1.f);
}
